// main_model_17635135717810
// MI455X (gfx1250) — compile-verified, matrix operand order checked by reading, not run on hardware
//
#include <hip/hip_runtime.h>
#include <hip/hip_bf16.h>

// ---------------------------------------------------------------------------
// Problem constants (from reference)
// ---------------------------------------------------------------------------
#define BATCH   256
#define NT      196     // tokens (14x14)
#define NTP     208     // padded to 13*16
#define DIM     256
#define HID     1536
#define HHEADS  8
#define KD      32
#define DV      128
#define DH      1024    // DV * H
#define KVP     224     // K-dim pad for P@V (7*32)
#define SST     209     // S row stride (floats) -> conflict free
#define PST     224     // P row stride (bf16)
#define NOFF    196     // unique (dx,dy) offsets for 14x14
#define MROWS   (BATCH*NT)   // 50176
#define ATTN_SCALE 0.1767766953f  // 32^-0.5

typedef __attribute__((ext_vector_type(16))) __bf16 v16bf;
typedef __attribute__((ext_vector_type(8)))  __bf16 v8bf;
typedef __attribute__((ext_vector_type(8)))  float  v8f;
typedef __attribute__((ext_vector_type(4)))  unsigned int v4u;
typedef __attribute__((ext_vector_type(4)))  int v4i;
typedef __attribute__((ext_vector_type(8)))  int v8i;

__device__ __forceinline__ v8f wmma_bf16(v16bf a, v16bf b, v8f c) {
  return __builtin_amdgcn_wmma_f32_16x16x32_bf16(false, a, false, b, (short)0, c,
                                                 false, false);
}

// Load one 16x32 bf16 operand fragment (A: MxK rows / B: NxK rows, K-contig).
__device__ __forceinline__ v16bf ld_frag(const __bf16* base, int stride, int lane) {
  return *(const v16bf*)(base + (size_t)(lane & 15) * stride + ((lane >> 4) << 4));
}

// ---------------------------------------------------------------------------
// TDM: 2-D bf16 tile DMA into LDS (tile = rows x 32 elems, row stride in elems)
// D# layout per CDNA5 ISA 8.3/8.4 (group0: count/lds/global/type, group1: dims)
// ---------------------------------------------------------------------------
#if __has_builtin(__builtin_amdgcn_tensor_load_to_lds)
#define HAVE_TDM 1
__device__ __forceinline__ void tdm_load_tile_bf16(unsigned lds_off,
                                                   const __bf16* gptr,
                                                   unsigned rows,
                                                   unsigned row_stride_elems) {
  unsigned long long ga = (unsigned long long)(const void*)gptr;
  v4u g0;
  g0[0] = 1u;                                               // count=1, no gather
  g0[1] = lds_off;                                          // lds_addr (bytes)
  g0[2] = (unsigned)ga;                                     // global_addr[31:0]
  g0[3] = (unsigned)((ga >> 32) & 0x01ffffffu) | (2u << 30);// addr[56:32]|type=2
  v8i g1;
  g1[0] = (int)(1u << 16);                                  // data_size=1 (2B)
  g1[1] = (int)(32u << 16);                                 // tensor_dim0 = 32
  g1[2] = (int)((rows & 0xffffu) << 16);                    // tensor_dim1 = rows
  g1[3] = (int)((rows >> 16) & 0xffffu) | (int)(32u << 16); // tile_dim0 = 32
  g1[4] = (int)(rows & 0xffffu);                            // tile_dim1 = rows
  g1[5] = (int)row_stride_elems;                            // dim0_stride[31:0]
  g1[6] = 0;
  g1[7] = 0;
  v4i zz = (v4i){0, 0, 0, 0};
#if __clang_major__ >= 23
  v8i z8 = (v8i){0, 0, 0, 0, 0, 0, 0, 0};
  __builtin_amdgcn_tensor_load_to_lds(g0, g1, zz, zz, z8, 0);
#else
  __builtin_amdgcn_tensor_load_to_lds(g0, g1, zz, zz, 0);
#endif
}
#else
#define HAVE_TDM 0
#endif

// ---------------------------------------------------------------------------
// f32 -> bf16 convert
// ---------------------------------------------------------------------------
__global__ void cvt_bf16_k(const float* __restrict__ in, __bf16* __restrict__ out,
                           size_t n) {
  size_t i = (size_t)blockIdx.x * blockDim.x + threadIdx.x;
  size_t stride = (size_t)gridDim.x * blockDim.x;
  for (; i < n; i += stride) out[i] = (__bf16)in[i];
}

// ---------------------------------------------------------------------------
// Expand relative-position bias to padded, mask-baked image:
// biasf[h][208][208];  (n,m) >= 196 -> -1e30 (softmax mask baked in)
// ---------------------------------------------------------------------------
__global__ void bias_expand_k(const float* __restrict__ btab,
                              const int* __restrict__ idxs,
                              float* __restrict__ biasf) {
  int idx = blockIdx.x * 256 + threadIdx.x;
  if (idx < HHEADS * NTP * NTP) {
    int hh = idx / (NTP * NTP);
    int rem = idx % (NTP * NTP);
    int n = rem / NTP, m = rem % NTP;
    float v = -1e30f;
    if (n < NT && m < NT) v = btab[hh * NOFF + idxs[n * NT + m]];
    biasf[idx] = v;
  }
}

// ---------------------------------------------------------------------------
// NT GEMM: C(MxN) = A(MxK bf16) * B(NxK bf16)^T. 128x128x32 tiles, 8 waves.
// TDM double-buffered tile DMA when available; plain LDS staging otherwise.
// ---------------------------------------------------------------------------
template <bool OUT_BF16>
__global__ void __launch_bounds__(256)
gemm_bf16_k(const __bf16* __restrict__ A, const __bf16* __restrict__ B,
            void* __restrict__ Cout, int M, int N, int K) {
  extern __shared__ char smem[];
  const int tid  = threadIdx.x;
  const int lane = tid & 31;
  const int wid  = tid >> 5;
  const int bm = blockIdx.x * 128;
  const int bn = blockIdx.y * 128;
  const int wm = (wid >> 1) * 32;
  const int wn = (wid & 1) * 64;

  v8f acc[2][4];
#pragma unroll
  for (int i = 0; i < 2; ++i)
#pragma unroll
    for (int j = 0; j < 4; ++j) acc[i][j] = (v8f){0.f,0.f,0.f,0.f,0.f,0.f,0.f,0.f};

  const size_t abase = (size_t)bm * K;
  const size_t bbase = (size_t)bn * K;

#if HAVE_TDM
  // Double-buffered TDM pipeline: buffer b at byte offset b*16384 (A 8KB + B 8KB)
  if (wid == 0) {
    tdm_load_tile_bf16(0u,    A + abase, 128, (unsigned)K);
    tdm_load_tile_bf16(8192u, B + bbase, 128, (unsigned)K);
  }
  int cur = 0;
  for (int k0 = 0; k0 < K; k0 += 32) {
    if (wid == 0) __builtin_amdgcn_s_wait_tensorcnt(0);
    __syncthreads();   // current buffer ready for all; prev reads done for all
    if (wid == 0 && k0 + 32 < K) {
      unsigned nb = (unsigned)((cur ^ 1) * 16384);
      tdm_load_tile_bf16(nb,        A + abase + k0 + 32, 128, (unsigned)K);
      tdm_load_tile_bf16(nb + 8192u, B + bbase + k0 + 32, 128, (unsigned)K);
    }
    const __bf16* As = (const __bf16*)(smem + cur * 16384);
    const __bf16* Bs = As + 4096;
    v16bf af[2], bf[4];
#pragma unroll
    for (int i = 0; i < 2; ++i) af[i] = ld_frag(As + (wm + i * 16) * 32, 32, lane);
#pragma unroll
    for (int j = 0; j < 4; ++j) bf[j] = ld_frag(Bs + (wn + j * 16) * 32, 32, lane);
#pragma unroll
    for (int i = 0; i < 2; ++i)
#pragma unroll
      for (int j = 0; j < 4; ++j) acc[i][j] = wmma_bf16(af[i], bf[j], acc[i][j]);
    cur ^= 1;
  }
#else
  __bf16* As = (__bf16*)smem;
  __bf16* Bs = As + 128 * 32;
  const int srow = tid >> 1;
  const int sseg = (tid & 1) << 4;
  for (int k0 = 0; k0 < K; k0 += 32) {
    const __bf16* ga = A + (size_t)(bm + srow) * K + k0 + sseg;
    const __bf16* gb = B + (size_t)(bn + srow) * K + k0 + sseg;
    *(v16bf*)(As + srow * 32 + sseg) = *(const v16bf*)ga;
    *(v16bf*)(Bs + srow * 32 + sseg) = *(const v16bf*)gb;
    if (k0 + 32 < K) {
      __builtin_prefetch(ga + 32, 0, 3);
      __builtin_prefetch(gb + 32, 0, 3);
    }
    __syncthreads();
    v16bf af[2], bf[4];
#pragma unroll
    for (int i = 0; i < 2; ++i) af[i] = ld_frag(As + (wm + i * 16) * 32, 32, lane);
#pragma unroll
    for (int j = 0; j < 4; ++j) bf[j] = ld_frag(Bs + (wn + j * 16) * 32, 32, lane);
#pragma unroll
    for (int i = 0; i < 2; ++i)
#pragma unroll
      for (int j = 0; j < 4; ++j) acc[i][j] = wmma_bf16(af[i], bf[j], acc[i][j]);
    __syncthreads();
  }
#endif

  const int nl = lane & 15;
  const int hi = (lane >> 4) * 8;
#pragma unroll
  for (int i = 0; i < 2; ++i)
#pragma unroll
    for (int j = 0; j < 4; ++j)
#pragma unroll
      for (int r = 0; r < 8; ++r) {
        int gm = bm + wm + i * 16 + r + hi;
        int gn = bn + wn + j * 16 + nl;
        float v = acc[i][j][r];
        if (OUT_BF16) ((__bf16*)Cout)[(size_t)gm * N + gn] = (__bf16)v;
        else          ((float*) Cout)[(size_t)gm * N + gn] = v;
      }
}

// ---------------------------------------------------------------------------
// Per-channel sum / sumsq accumulation
// ---------------------------------------------------------------------------
__global__ void __launch_bounds__(256)
stats_k(const void* __restrict__ data, int is_bf16, int C, int cpt,
        float* __restrict__ sum, float* __restrict__ sumsq) {
  const int tid = threadIdx.x;
  const size_t r0 = (size_t)blockIdx.x * 256;
  float sa[6], qa[6];
  for (int j = 0; j < cpt; ++j) { sa[j] = 0.f; qa[j] = 0.f; }
  for (int rr = 0; rr < 256; ++rr) {
    size_t row = r0 + rr;
    for (int j = 0; j < cpt; ++j) {
      int c = tid + j * 256;
      float v = is_bf16 ? (float)((const __bf16*)data)[row * C + c]
                        : ((const float*)data)[row * C + c];
      sa[j] += v;
      qa[j] += v * v;
    }
  }
  for (int j = 0; j < cpt; ++j) {
    int c = tid + j * 256;
    atomicAdd(&sum[c], sa[j]);
    atomicAdd(&sumsq[c], qa[j]);
  }
}

__global__ void bnfin_k(const float* __restrict__ sum, const float* __restrict__ sq,
                        const float* __restrict__ g, const float* __restrict__ b,
                        float invcnt, float* __restrict__ scale,
                        float* __restrict__ shift, int C) {
  int c = blockIdx.x * 256 + threadIdx.x;
  if (c < C) {
    float mu  = sum[c] * invcnt;
    float var = sq[c] * invcnt - mu * mu;
    float sc  = g[c] * rsqrtf(var + 1e-5f);
    scale[c] = sc;
    shift[c] = b[c] - mu * sc;
  }
}

// ---------------------------------------------------------------------------
// Attention: one workgroup per (batch, head).
// ---------------------------------------------------------------------------
__global__ void __launch_bounds__(256)
attn_k(const __bf16* __restrict__ hmat, const float* __restrict__ scale1,
       const float* __restrict__ shift1, const float* __restrict__ biasf,
       __bf16* __restrict__ obuf) {
  extern __shared__ char smem[];
  __bf16* qs  = (__bf16*)smem;              // NTP x 32
  __bf16* ks  = qs + NTP * 32;              // NTP x 32
  __bf16* vsT = ks + NTP * 32;              // DV x KVP (v^T, K-contig)
  float*  S   = (float*)(vsT + DV * KVP);   // 64 x SST
  __bf16* P   = (__bf16*)(S + 64 * SST);    // 64 x PST
  float*  sc  = (float*)(P + 64 * PST);     // 192
  float*  sh  = sc + 192;                   // 192

  const int tid  = threadIdx.x;
  const int lane = tid & 31;
  const int wid  = tid >> 5;
  const int b  = blockIdx.x >> 3;
  const int hh = blockIdx.x & 7;
  const size_t hbase = ((size_t)b * NT) * HID + hh * 192;
  const float* brow = biasf + (size_t)hh * NTP * NTP;

  // per-head BN params to LDS
  for (int i = tid; i < 192; i += 256) {
    sc[i] = scale1[hh * 192 + i];
    sh[i] = shift1[hh * 192 + i];
  }
  __syncthreads();

  // ---- stage q,k (vector loads, BN applied, zero-padded rows) ----
  for (int idx = tid; idx < NTP * 4; idx += 256) {
    int n = idx >> 2, c8 = (idx & 3) << 3;
    v8bf qo, ko;
    if (n < NT) {
      const __bf16* rp = hmat + hbase + (size_t)n * HID;
      v8bf qr = *(const v8bf*)(rp + c8);
      v8bf kr = *(const v8bf*)(rp + 32 + c8);
#pragma unroll
      for (int e = 0; e < 8; ++e) {
        int cc = c8 + e;
        qo[e] = (__bf16)((float)qr[e] * sc[cc]      + sh[cc]);
        ko[e] = (__bf16)((float)kr[e] * sc[32 + cc] + sh[32 + cc]);
      }
    } else {
#pragma unroll
      for (int e = 0; e < 8; ++e) { qo[e] = (__bf16)0.f; ko[e] = (__bf16)0.f; }
    }
    *(v8bf*)(qs + n * 32 + c8) = qo;
    *(v8bf*)(ks + n * 32 + c8) = ko;
  }
  // ---- stage v transposed ----
  for (int idx = tid; idx < KVP * 16; idx += 256) {
    int n = idx >> 4, c8 = (idx & 15) << 3;
    if (n < NT) {
      const __bf16* rp = hmat + hbase + (size_t)n * HID + 64;
      v8bf vr = *(const v8bf*)(rp + c8);
#pragma unroll
      for (int e = 0; e < 8; ++e) {
        int cc = c8 + e;
        vsT[cc * KVP + n] = (__bf16)((float)vr[e] * sc[64 + cc] + sh[64 + cc]);
      }
    } else {
#pragma unroll
      for (int e = 0; e < 8; ++e) vsT[(c8 + e) * KVP + n] = (__bf16)0.f;
    }
  }
  __syncthreads();

  for (int mt0 = 0; mt0 < 13; mt0 += 4) {
    const int mts   = (13 - mt0 < 4) ? (13 - mt0) : 4;
    const int rows0 = mt0 * 16;

    // ---- S = q k^T * scale + bias (mask baked into biasf; no branches) ----
    for (int t = wid; t < mts * 13; t += 8) {
      int i = t / 13, j = t % 13;
      v16bf a  = ld_frag(qs + (rows0 + i * 16) * 32, 32, lane);
      v16bf bb = ld_frag(ks + (j * 16) * 32, 32, lane);
      v8f cacc = (v8f){0.f,0.f,0.f,0.f,0.f,0.f,0.f,0.f};
      cacc = wmma_bf16(a, bb, cacc);
      int nl = lane & 15, hi8 = (lane >> 4) * 8;
#pragma unroll
      for (int r = 0; r < 8; ++r) {
        int lm = i * 16 + r + hi8;
        int gm = rows0 + lm;
        int gn = j * 16 + nl;
        S[lm * SST + gn] = cacc[r] * ATTN_SCALE + brow[gm * NTP + gn];
      }
    }
    __syncthreads();

    // ---- row softmax -> P (bf16); pad cols vanish via exp(-1e30) ----
    if (tid < mts * 16 && rows0 + tid < NT) {
      float* Sr = S + tid * SST;
      float m = -1e30f;
      for (int c = 0; c < NTP; ++c) m = fmaxf(m, Sr[c]);
      float sum = 0.f;
      for (int c = 0; c < NTP; ++c) { float e = __expf(Sr[c] - m); Sr[c] = e; sum += e; }
      float inv = 1.f / sum;
      __bf16* Pr = P + tid * PST;
      for (int c = 0; c < NTP; ++c)   Pr[c] = (__bf16)(Sr[c] * inv);
      for (int c = NTP; c < KVP; ++c) Pr[c] = (__bf16)0.f;
    }
    __syncthreads();

    // ---- O = P @ V (7 chained WMMAs per tile), hardswish fused ----
    for (int t = wid; t < mts * 8; t += 8) {
      int i = t >> 3, jo = t & 7;
      v8f cacc = (v8f){0.f,0.f,0.f,0.f,0.f,0.f,0.f,0.f};
#pragma unroll
      for (int kk = 0; kk < KVP / 32; ++kk) {
        v16bf a  = ld_frag(P + i * 16 * PST + kk * 32, PST, lane);
        v16bf bb = ld_frag(vsT + (jo * 16) * KVP + kk * 32, KVP, lane);
        cacc = wmma_bf16(a, bb, cacc);
      }
      int nl = lane & 15, hi8 = (lane >> 4) * 8;
#pragma unroll
      for (int r = 0; r < 8; ++r) {
        int gm = rows0 + i * 16 + r + hi8;
        if (gm < NT) {
          float v = cacc[r];
          float hsw = v * fminf(fmaxf(v + 3.f, 0.f), 6.f) * (1.f / 6.f);
          int gc = hh * DV + jo * 16 + nl;
          obuf[((size_t)b * NT + gm) * DH + gc] = (__bf16)hsw;
        }
      }
    }
    __syncthreads();
  }
}

// ---------------------------------------------------------------------------
// Final BN apply
// ---------------------------------------------------------------------------
__global__ void outnorm_k(const float* __restrict__ y, const float* __restrict__ scale,
                          const float* __restrict__ shift, float* __restrict__ out,
                          size_t n) {
  size_t i = (size_t)blockIdx.x * blockDim.x + threadIdx.x;
  size_t stride = (size_t)gridDim.x * blockDim.x;
  for (; i < n; i += stride) {
    int c = (int)(i & 255);
    out[i] = y[i] * scale[c] + shift[c];
  }
}

// ---------------------------------------------------------------------------
// Host launcher
// ---------------------------------------------------------------------------
extern "C" void kernel_launch(void* const* d_in, const int* in_sizes, int n_in,
                              void* d_out, int out_size, void* d_ws, size_t ws_size,
                              hipStream_t stream) {
  const float* x     = (const float*)d_in[0];
  const float* Wqkv  = (const float*)d_in[1];
  const float* g1    = (const float*)d_in[2];
  const float* b1    = (const float*)d_in[3];
  const float* btab  = (const float*)d_in[4];
  const float* Wproj = (const float*)d_in[5];
  const float* g2    = (const float*)d_in[6];
  const float* b2    = (const float*)d_in[7];
  const int*   idxs  = (const int*)d_in[8];
  float* out = (float*)d_out;

  char* ws = (char*)d_ws;
  size_t off = 0;
  __bf16* xb  = (__bf16*)(ws + off); off += (size_t)MROWS * DIM * 2;
  __bf16* w1b = (__bf16*)(ws + off); off += (size_t)HID * DIM * 2;
  __bf16* wpb = (__bf16*)(ws + off); off += (size_t)DIM * DH * 2;
  __bf16* hb  = (__bf16*)(ws + off); off += (size_t)MROWS * HID * 2;
  __bf16* ob  = (__bf16*)(ws + off); off += (size_t)MROWS * DH * 2;
  float*  yb  = (float*)(ws + off);  off += (size_t)MROWS * DIM * 4;
  float*  stats = (float*)(ws + off);
  float* sum1 = stats;            float* sq1 = stats + HID;
  float* sum2 = stats + 2 * HID;  float* sq2 = stats + 2 * HID + DIM;
  off += (size_t)(2 * HID + 2 * DIM) * 4;
  float* scale1 = (float*)(ws + off); off += HID * 4;
  float* shift1 = (float*)(ws + off); off += HID * 4;
  float* scale2 = (float*)(ws + off); off += DIM * 4;
  float* shift2 = (float*)(ws + off); off += DIM * 4;
  float* biasf  = (float*)(ws + off); off += (size_t)HHEADS * NTP * NTP * 4;

  hipMemsetAsync(stats, 0, (size_t)(2 * HID + 2 * DIM) * 4, stream);

  // converts + bias expansion (independent prep work)
  cvt_bf16_k<<<4096, 256, 0, stream>>>(x, xb, (size_t)MROWS * DIM);
  cvt_bf16_k<<<512, 256, 0, stream>>>(Wqkv, w1b, (size_t)HID * DIM);
  cvt_bf16_k<<<512, 256, 0, stream>>>(Wproj, wpb, (size_t)DIM * DH);
  bias_expand_k<<<(HHEADS * NTP * NTP + 255) / 256, 256, 0, stream>>>(btab, idxs, biasf);

  // QKV GEMM: h(50176x1536) = x * Wqkv^T   (TDM double-buffer: 32 KB LDS)
  {
    dim3 grid(MROWS / 128, HID / 128);
    gemm_bf16_k<true><<<grid, 256, 32768, stream>>>(xb, w1b, (void*)hb,
                                                    MROWS, HID, DIM);
  }

  // BN1
  stats_k<<<MROWS / 256, 256, 0, stream>>>(hb, 1, HID, HID / 256, sum1, sq1);
  bnfin_k<<<(HID + 255) / 256, 256, 0, stream>>>(sum1, sq1, g1, b1,
                                                 1.f / (float)MROWS, scale1, shift1, HID);

  // attention
  {
    size_t lds = (size_t)NTP * 32 * 2 * 2 + (size_t)DV * KVP * 2
               + (size_t)64 * SST * 4 + (size_t)64 * PST * 2
               + 2 * 192 * 4;                 // == 167680 bytes
    attn_k<<<BATCH * HHEADS, 256, lds, stream>>>(hb, scale1, shift1, biasf, ob);
  }

  // proj GEMM: y(50176x256) = o(50176x1024) * Wproj^T
  {
    dim3 grid(MROWS / 128, DIM / 128);
    gemm_bf16_k<false><<<grid, 256, 32768, stream>>>(ob, wpb, (void*)yb,
                                                     MROWS, DIM, DH);
  }

  // BN2 + apply
  stats_k<<<MROWS / 256, 256, 0, stream>>>(yb, 0, DIM, DIM / 256, sum2, sq2);
  bnfin_k<<<(DIM + 255) / 256, 256, 0, stream>>>(sum2, sq2, g2, b2,
                                                 1.f / (float)MROWS, scale2, shift2, DIM);
  outnorm_k<<<4096, 256, 0, stream>>>(yb, scale2, shift2, out, (size_t)MROWS * DIM);
}
